// MultiHeadAttention_70540542870156
// MI455X (gfx1250) — compile-verified, matrix operand order checked by reading, not run on hardware
//
#include <hip/hip_runtime.h>

// MHA forward for B=2, L=2048, D=1024, H=16 on gfx1250 (MI455X).
// All matmuls via v_wmma_f32_16x16x32_f16 (f16 in, f32 accumulate).
// - Weights are pre-transposed+converted to f16 [N][K] so every WMMA B tile
//   is a contiguous 16B chunk (global_load_b128), same as the A side.
// - V projection stores its output pre-transposed per head (Vt[b][h][dk][L])
//   so the P@V B-operand loads are contiguous too.
// - Attention probabilities are computed, softmaxed, and consumed (P@V)
//   entirely in LDS (16x2048 f32 stripe = 128KB of the 320KB WGP LDS); the
//   536MB attn output is written to HBM exactly once (~26us at 23.3 TB/s).
// Workspace: Qh,Kh,Vt,ctx (4 x 8MiB f16) + 4 transposed weights (4 x 2MiB).

typedef __attribute__((ext_vector_type(16))) _Float16 v16h;
typedef __attribute__((ext_vector_type(8)))  float    v8f;
typedef _Float16 half_t;

constexpr int kB  = 2;
constexpr int kL  = 2048;
constexpr int kD  = 1024;
constexpr int kH  = 16;
constexpr int kDK = 64;
constexpr int kM  = kB * kL;   // 4096 rows in the projection GEMMs

static __device__ inline v8f wmma_f16(v16h a, v16h b, v8f c) {
  // 8 args: (neg_a, A, neg_b, B, c_mod, C, reuse_a, reuse_b)
  return __builtin_amdgcn_wmma_f32_16x16x32_f16(false, a, false, b, (short)0, c,
                                                false, false);
}

// Operand chunk (16 elements along K) per ISA 7.12.2: lane holds one row/col;
// lanes 0-15 carry K = {k..k+7, k+16..k+23}, lanes 16-31 carry
// K = {k+8..k+15, k+24..k+31}. p = row/col base pointer, k includes the
// per-lane (lane>>4)<<3 offset. Contiguous in memory -> b128 loads for f16.
template <typename T>
static __device__ inline v16h load_chunk(const T* __restrict__ p, int k) {
  v16h a;
#pragma unroll
  for (int j = 0; j < 8; ++j) {
    a[j]     = (_Float16)p[k + j];
    a[j + 8] = (_Float16)p[k + 16 + j];
  }
  return a;
}

// ---------------------------------------------------------------------------
// Wt[n][k] = (f16) W[k][n].  LDS 32x32 tile transpose, both sides coalesced.
// ---------------------------------------------------------------------------
__global__ __launch_bounds__(256) void transpose_w_kernel(
    const float* __restrict__ W, half_t* __restrict__ Wt, int N, int K) {
  __shared__ float tile[32][33];
  const int nbx = N >> 5;
  const int bx  = blockIdx.x % nbx;   // column tile of W
  const int by  = blockIdx.x / nbx;   // row tile of W
  const int tx  = threadIdx.x & 31;
  const int ty  = threadIdx.x >> 5;   // 0..7
#pragma unroll
  for (int i = 0; i < 4; ++i)
    tile[ty + i * 8][tx] = W[(size_t)(by * 32 + ty + i * 8) * N + bx * 32 + tx];
  __syncthreads();
#pragma unroll
  for (int i = 0; i < 4; ++i)
    Wt[(size_t)(bx * 32 + ty + i * 8) * K + by * 32 + tx] =
        (half_t)tile[tx][ty + i * 8];
}

// ---------------------------------------------------------------------------
// Y = X @ W + bias.  X: [M,K] f32 or f16.  Wt: f16 [N][K] (pre-transposed).
// Per wave: 32x32 output (2x2 WMMA tiles). 8 waves -> 64x128 per workgroup.
// mode 0: Y f16 [M][N] row-major (Q/K projections)
// mode 1: Y f32 [M][N] row-major + bias (output projection)
// mode 2: Y f16 head-transposed Vt[((b*H+h)*DK+dk)*L + q] (V projection)
// ---------------------------------------------------------------------------
template <typename AT>
__global__ __launch_bounds__(256) void gemm_xw_kernel(
    const AT* __restrict__ X, const half_t* __restrict__ Wt,
    const float* __restrict__ bias, half_t* __restrict__ Yh,
    float* __restrict__ Yf, int N, int K, int mode) {
  const int tid  = threadIdx.x;
  const int lane = tid & 31;
  const int wv   = tid >> 5;
  const int wr   = wv >> 2;            // 0..1
  const int wc   = wv & 3;             // 0..3
  const int nbn  = N >> 7;             // column blocks of 128
  const int bm   = blockIdx.x / nbn;
  const int bn   = blockIdx.x % nbn;
  const int row0 = bm * 64 + wr * 32;
  const int col0 = bn * 128 + wc * 32;

  const int ln   = lane & 15;
  const int koff = (lane >> 4) << 3;   // 0 or 8
  const int rb   = (lane >> 4) << 3;   // C row offset per half-wave

  const AT* arow0 = X + (size_t)(row0 + ln) * K;
  const AT* arow1 = arow0 + (size_t)16 * K;
  const half_t* brow0 = Wt + (size_t)(col0 + ln) * K;   // contiguous along K
  const half_t* brow1 = brow0 + (size_t)16 * K;

  v8f c00 = {}, c01 = {}, c10 = {}, c11 = {};
  for (int k0 = 0; k0 < K; k0 += 32) {
    __builtin_prefetch(arow0 + k0 + 64, 0, 1);   // global_prefetch_b8
    v16h a0 = load_chunk(arow0, k0 + koff);
    v16h a1 = load_chunk(arow1, k0 + koff);
    v16h b0 = load_chunk(brow0, k0 + koff);
    v16h b1 = load_chunk(brow1, k0 + koff);
    c00 = wmma_f16(a0, b0, c00);
    c01 = wmma_f16(a0, b1, c01);
    c10 = wmma_f16(a1, b0, c10);
    c11 = wmma_f16(a1, b1, c11);
  }

  const float bias0 = bias[col0 + ln];
  const float bias1 = bias[col0 + 16 + ln];
#pragma unroll
  for (int v = 0; v < 8; ++v) {
    const int rA = row0 + rb + v;
    const int rB = rA + 16;
    const int n0 = col0 + ln;
    const int n1 = n0 + 16;
    const float v00 = c00[v] + bias0, v01 = c01[v] + bias1;
    const float v10 = c10[v] + bias0, v11 = c11[v] + bias1;
    if (mode == 0) {
      Yh[(size_t)rA * N + n0] = (half_t)v00;
      Yh[(size_t)rA * N + n1] = (half_t)v01;
      Yh[(size_t)rB * N + n0] = (half_t)v10;
      Yh[(size_t)rB * N + n1] = (half_t)v11;
    } else if (mode == 1) {
      Yf[(size_t)rA * N + n0] = v00;
      Yf[(size_t)rA * N + n1] = v01;
      Yf[(size_t)rB * N + n0] = v10;
      Yf[(size_t)rB * N + n1] = v11;
    } else {  // mode 2: Vt[((b*H + h)*DK + dk)*L + q]
      auto storeT = [&](int row, int col, float val) {
        const int bb = row >> 11, qq = row & (kL - 1);
        const int hh = col >> 6,  dk = col & (kDK - 1);
        Yh[(((size_t)bb * kH + hh) * kDK + dk) * kL + qq] = (half_t)val;
      };
      storeT(rA, n0, v00);
      storeT(rA, n1, v01);
      storeT(rB, n0, v10);
      storeT(rB, n1, v11);
    }
  }
}

// ---------------------------------------------------------------------------
// Fused attention: one workgroup per (b, h, 16 q-rows).
//  Phase 1: S = (Qh Kh^T)/D, masked, into LDS (16 x 2048 f32).
//  Phase 2: row softmax in LDS, single coalesced write of P to global.
//  Phase 3: ctx = P @ V from LDS (V pre-transposed -> contiguous B loads),
//           K split over wave pairs + small LDS reduce.
// Dynamic LDS: 16*2048 + 2*16*64 floats = 136 KiB.
// ---------------------------------------------------------------------------
__global__ __launch_bounds__(256) void fused_attn_kernel(
    const half_t* __restrict__ Qh, const half_t* __restrict__ Kh,
    const half_t* __restrict__ Vt, const int* __restrict__ mask,
    float* __restrict__ attnOut, half_t* __restrict__ ctx) {
  extern __shared__ float smem[];
  float* sP   = smem;               // [16][2048] scores -> probabilities
  float* sRed = smem + 16 * kL;     // [2][16][64] partial ctx

  const int tid  = threadIdx.x;
  const int lane = tid & 31;
  const int wv   = tid >> 5;        // 0..7
  int blk = blockIdx.x;
  const int qt = blk & 127; blk >>= 7;   // kL/16 q-tiles
  const int h  = blk & 15;  blk >>= 4;
  const int b  = blk;                    // 0..1
  const int row0 = qt * 16;
  const int ln   = lane & 15;
  const int koff = (lane >> 4) << 3;
  const int rb   = (lane >> 4) << 3;

  // ---- Phase 1: masked scaled scores into LDS ----
  const half_t* arow = Qh + (size_t)(b * kL + row0 + ln) * kD + h * kDK;
  const v16h a_lo = load_chunk(arow, koff);        // dk 0..31
  const v16h a_hi = load_chunk(arow, 32 + koff);   // dk 32..63
  const float scale = 1.0f / (float)kD;

  for (int t = 0; t < 16; ++t) {            // 16 col-tiles = 256 cols per wave
    const int col0 = wv * 256 + t * 16;
    const half_t* bcol = Kh + (size_t)(b * kL + col0 + ln) * kD + h * kDK;
    v16h b_lo = load_chunk(bcol, koff);     // contiguous along dk -> b128
    v16h b_hi = load_chunk(bcol, 32 + koff);
    v8f acc = {};
    acc = wmma_f16(a_lo, b_lo, acc);
    acc = wmma_f16(a_hi, b_hi, acc);
    const int colw = col0 + ln;
#pragma unroll
    for (int v = 0; v < 8; ++v) {
      const int q = rb + v;                 // 0..15 within the tile
      float s = acc[v] * scale;
      if (mask[((size_t)(b * kL + row0 + q)) * kL + colw] == 0) s = -1e9f;
      sP[q * kL + colw] = s;
    }
  }
  __syncthreads();

  // ---- Phase 2: softmax per row (2 rows per wave), write P once ----
#pragma unroll
  for (int rr = 0; rr < 2; ++rr) {
    const int r = wv * 2 + rr;
    float* prow = sP + r * kL;
    float mx = -3.0e38f;
    for (int i = lane; i < kL; i += 32) mx = fmaxf(mx, prow[i]);
#pragma unroll
    for (int off = 16; off > 0; off >>= 1)
      mx = fmaxf(mx, __shfl_xor(mx, off, 32));
    float sum = 0.0f;
    for (int i = lane; i < kL; i += 32) {
      float e = __expf(prow[i] - mx);
      prow[i] = e;
      sum += e;
    }
#pragma unroll
    for (int off = 16; off > 0; off >>= 1) sum += __shfl_xor(sum, off, 32);
    const float inv = 1.0f / sum;
    float* grow = attnOut + (((size_t)(h * kB + b)) * kL + row0 + r) * kL;
    for (int i = lane; i < kL; i += 32) {
      const float p = prow[i] * inv;
      prow[i] = p;        // keep in LDS for phase 3
      grow[i] = p;        // the single mandatory HBM write of attn
    }
  }
  __syncthreads();

  // ---- Phase 3: ctx = P @ V. 8 waves = 4 dk-tiles x 2 K-halves ----
  const int t3  = wv & 3;             // dk tile: cols t3*16 .. +15
  const int kh3 = wv >> 2;            // K half: [kh3*1024, +1024)
  const float*  par  = sP + (size_t)ln * kL;   // P row ln (LDS)
  const half_t* vrow =                          // V^T row (dk) -> contiguous k
      Vt + (((size_t)(b * kH + h)) * kDK + t3 * 16 + ln) * kL;
  v8f acc = {};
  const int kbeg = kh3 * 1024;
  for (int k0 = kbeg; k0 < kbeg + 1024; k0 += 32) {
    v16h a  = load_chunk(par, k0 + koff);
    v16h bt = load_chunk(vrow, k0 + koff);
    acc = wmma_f16(a, bt, acc);
  }
#pragma unroll
  for (int v = 0; v < 8; ++v)
    sRed[(kh3 * 16 + rb + v) * 64 + t3 * 16 + ln] = acc[v];
  __syncthreads();

  // reduce the two K-halves; 1024 outputs over 256 threads
#pragma unroll
  for (int i = 0; i < 4; ++i) {
    const int idx = tid * 4 + i;        // 0..1023
    const int q   = idx >> 6;           // 0..15
    const int dk  = idx & 63;
    const float s = sRed[q * 64 + dk] + sRed[(16 + q) * 64 + dk];
    ctx[((size_t)(b * kL) + row0 + q) * kD + h * kDK + dk] = (half_t)s;
  }
}

// ---------------------------------------------------------------------------
extern "C" void kernel_launch(void* const* d_in, const int* in_sizes, int n_in,
                              void* d_out, int out_size, void* d_ws,
                              size_t ws_size, hipStream_t stream) {
  const float* q    = (const float*)d_in[0];
  const float* k    = (const float*)d_in[1];
  const float* v    = (const float*)d_in[2];
  const int*   mask = (const int*)d_in[3];
  const float* Wq   = (const float*)d_in[4];
  const float* bq   = (const float*)d_in[5];
  const float* Wk   = (const float*)d_in[6];
  const float* bk   = (const float*)d_in[7];
  const float* Wv   = (const float*)d_in[8];
  const float* bv   = (const float*)d_in[9];
  const float* Wo   = (const float*)d_in[10];
  const float* bo   = (const float*)d_in[11];

  float* out  = (float*)d_out;
  float* attn = out + (size_t)kM * kD;   // tuple output #2, [H*B, L, L]

  const size_t nMD = (size_t)kM * kD;    // 4M elements
  const size_t nW  = (size_t)kD * kD;    // 1M elements per weight
  half_t* qh  = (half_t*)d_ws;
  half_t* kh  = qh + nMD;
  half_t* vt  = kh + nMD;
  half_t* ctx = vt + nMD;
  half_t* wqT = ctx + nMD;
  half_t* wkT = wqT + nW;
  half_t* wvT = wkT + nW;
  half_t* woT = wvT + nW;

  const dim3 blk(256);

  // one-time weight transpose+convert: f32 [K][N] -> f16 [N][K]
  const int trGrid = (kD / 32) * (kD / 32);     // 1024
  transpose_w_kernel<<<trGrid, blk, 0, stream>>>(Wq, wqT, kD, kD);
  transpose_w_kernel<<<trGrid, blk, 0, stream>>>(Wk, wkT, kD, kD);
  transpose_w_kernel<<<trGrid, blk, 0, stream>>>(Wv, wvT, kD, kD);
  transpose_w_kernel<<<trGrid, blk, 0, stream>>>(Wo, woT, kD, kD);

  const int gemmGrid = (kM / 64) * (kD / 128);  // 512
  gemm_xw_kernel<float><<<gemmGrid, blk, 0, stream>>>(
      q, wqT, bq, qh, (float*)nullptr, kD, kD, /*mode=*/0);
  gemm_xw_kernel<float><<<gemmGrid, blk, 0, stream>>>(
      k, wkT, bk, kh, (float*)nullptr, kD, kD, /*mode=*/0);
  gemm_xw_kernel<float><<<gemmGrid, blk, 0, stream>>>(
      v, wvT, bv, vt, (float*)nullptr, kD, kD, /*mode=*/2);

  const int attnGrid = kB * kH * (kL / 16);     // 4096
  const size_t ldsBytes = (size_t)(16 * kL + 2 * 16 * 64) * sizeof(float);
  fused_attn_kernel<<<attnGrid, blk, ldsBytes, stream>>>(qh, kh, vt, mask,
                                                         attn, ctx);

  gemm_xw_kernel<half_t><<<gemmGrid, blk, 0, stream>>>(
      ctx, woT, bo, (half_t*)nullptr, out, kD, kD, /*mode=*/1);
}
